// MultiHeadAttention_39548058861924
// MI455X (gfx1250) — compile-verified, matrix operand order checked by reading, not run on hardware
//
#include <hip/hip_runtime.h>
#include <hip/hip_bf16.h>

#define Bz   4
#define Sz   2048
#define DIN  1024
#define DOUT 1024
#define Hh   16
#define HD   64
#define MROWS (Bz * Sz)   // 8192

typedef __attribute__((ext_vector_type(16))) __bf16 v16bf;
typedef __attribute__((ext_vector_type(8)))  __bf16 v8bf;
typedef __attribute__((ext_vector_type(8)))  float  v8f;
typedef __attribute__((ext_vector_type(4)))  unsigned int v4u;
typedef __attribute__((ext_vector_type(8)))  int v8i_;
typedef __attribute__((ext_vector_type(4)))  int v4i_;

#ifndef __has_builtin
#define __has_builtin(x) 0
#endif
#if defined(__HIP_DEVICE_COMPILE__) && __has_builtin(__builtin_amdgcn_tensor_load_to_lds) && __has_builtin(__builtin_amdgcn_s_wait_tensorcnt)
#define HAVE_TDM 1
#else
#define HAVE_TDM 0
#endif

// ---------------------------------------------------------------------------
// A/B fragment loader for v_wmma_f32_16x16x32_bf16.
// Per ISA 16-bit A 16x32 layout: lane l (row = l%16, half = l/16) holds
//   elements e=0..7  -> K = kbase + 8*half + e        (contiguous 8 bf16)
//   elements e=8..15 -> K = kbase + 16 + 8*half + e-8 (contiguous 8 bf16)
// B is loaded as B^T (N x K) with the identical per-lane map (N = lane%16).
// Works for both global and LDS (generic) pointers.
// ---------------------------------------------------------------------------
__device__ __forceinline__ v16bf load_frag(const __bf16* row_ptr, int kbase, int lh) {
    const v8bf* p0 = (const v8bf*)(row_ptr + kbase + (lh << 3));
    const v8bf* p1 = (const v8bf*)(row_ptr + kbase + 16 + (lh << 3));
    v8bf a = *p0;
    v8bf b = *p1;
    v16bf r;
#pragma unroll
    for (int i = 0; i < 8; ++i) { r[i] = a[i]; r[i + 8] = b[i]; }
    return r;
}

__device__ __forceinline__ v8f wmma_bf16(v16bf a, v16bf b, v8f c) {
    return __builtin_amdgcn_wmma_f32_16x16x32_bf16(false, a, false, b, (short)0, c, false, false);
}

#if HAVE_TDM
// ---------------------------------------------------------------------------
// TDM: async DMA of a 32(key) x 64(hd) bf16 tile (row stride 64 elements,
// i.e. fully contiguous 4KB) from global K into LDS.  D# packing per ISA
// cdna5 §8.3/8.4.  Wave-uniform operands forced scalar via readfirstlane.
// ---------------------------------------------------------------------------
__device__ __forceinline__ void tdm_load_k_tile(unsigned lds_addr, unsigned long long ga) {
    unsigned ga_lo = (unsigned)__builtin_amdgcn_readfirstlane((int)(unsigned)(ga & 0xFFFFFFFFull));
    unsigned ga_hi = (unsigned)__builtin_amdgcn_readfirstlane((int)(unsigned)(ga >> 32));
    unsigned la    = (unsigned)__builtin_amdgcn_readfirstlane((int)lds_addr);
    // group0: count=1 | lds_addr | global_addr[31:0] | global_addr[56:32], type=2
    v4u g0 = { 1u, la, ga_lo, (ga_hi & 0x01FFFFFFu) | (2u << 30) };
    // group1: data_size=2B; tensor_dim0=64, tensor_dim1=2048; tile 64x32;
    //         tensor_dim0_stride=64 elements; everything else 0.
    v8i_ g1 = { 0x00010000,                 // wg_mask=0, data_size=1(2B)
                (int)((64u & 0xFFFFu) << 16),   // tensor_dim0[15:0] << 16
                (int)((2048u & 0xFFFFu) << 16), // tensor_dim0[31:16]=0 | tensor_dim1[15:0]<<16
                (int)((64u & 0xFFFFu) << 16),   // tensor_dim1[31:16]=0 | tile_dim0=64 <<16
                32,                              // tile_dim1=32 | tile_dim2=0
                64,                              // tensor_dim0_stride[31:0]
                0,                               // stride0[47:32] | stride1[15:0]<<16
                0 };                             // stride1[47:16]
    v4i_ gz = { 0, 0, 0, 0 };
#if defined(__clang_major__) && (__clang_major__ >= 23)
    v8i_ gz8 = { 0, 0, 0, 0, 0, 0, 0, 0 };
    __builtin_amdgcn_tensor_load_to_lds(g0, g1, gz, gz, gz8, 0);
#else
    __builtin_amdgcn_tensor_load_to_lds(g0, g1, gz, gz, 0);
#endif
}
#endif

// ---------------------------------------------------------------------------
// Kernel 0a: convert x (f32) -> bf16
// ---------------------------------------------------------------------------
__global__ void cvt_x_kernel(const float* __restrict__ x, __bf16* __restrict__ xb, int n) {
    int i = blockIdx.x * 256 + threadIdx.x;
    if (i < n) xb[i] = (__bf16)x[i];
}

// ---------------------------------------------------------------------------
// Kernel 0b: convert + transpose weights: Wt[n][k] = bf16(W[k][n])
// ---------------------------------------------------------------------------
__global__ void cvt_wt_kernel(const float* __restrict__ Wq, const float* __restrict__ Wk,
                              const float* __restrict__ Wv, const float* __restrict__ Wo,
                              __bf16* __restrict__ Wqt, __bf16* __restrict__ Wkt,
                              __bf16* __restrict__ Wvt, __bf16* __restrict__ Wot) {
    int i = blockIdx.x * 256 + threadIdx.x;      // i over n*1024 + k
    int z = blockIdx.z;
    const float* W  = (z == 0) ? Wq  : (z == 1) ? Wk  : (z == 2) ? Wv  : Wo;
    __bf16*      Wt = (z == 0) ? Wqt : (z == 1) ? Wkt : (z == 2) ? Wvt : Wot;
    int n = i >> 10;
    int k = i & 1023;
    Wt[i] = (__bf16)W[(size_t)k * DOUT + n];
}

// ---------------------------------------------------------------------------
// Kernel 1: fused QKV projection.  out = x @ W  (M=8192, N=1024, K=1024).
// 32(M) x 64(N) register tile per wave: 2 A-frags x 4 B-frags -> 8 WMMAs per
// 32-wide K-step at only 12 b128 loads (1.5 loads/WMMA).
// Q,K stored (B,H,S,HD) bf16; V stored head-transposed (B,H,HD,S) bf16.
// ---------------------------------------------------------------------------
__global__ __launch_bounds__(128) void proj_qkv_kernel(
        const __bf16* __restrict__ xb,
        const __bf16* __restrict__ Wqt, const __bf16* __restrict__ Wkt,
        const __bf16* __restrict__ Wvt,
        __bf16* __restrict__ Qb, __bf16* __restrict__ Kb, __bf16* __restrict__ Vtb) {
    const int wave  = threadIdx.x >> 5;
    const int lane  = threadIdx.x & 31;
    const int row16 = lane & 15;
    const int lh    = lane >> 4;

    const int m0    = blockIdx.x * 32;                  // row tile base (b*S+s)
    const int n0    = blockIdx.y * 256 + wave * 64;     // col tile base (d_out)
    const int which = blockIdx.z;                       // 0=Q 1=K 2=V

    const __bf16* Wt = (which == 0) ? Wqt : (which == 1) ? Wkt : Wvt;
    const __bf16* arow0 = xb + (size_t)(m0 + row16) * DIN;
    const __bf16* arow1 = arow0 + (size_t)16 * DIN;
    const __bf16* brow[4];
#pragma unroll
    for (int j = 0; j < 4; ++j) brow[j] = Wt + (size_t)(n0 + 16 * j + row16) * DIN;

    v8f acc[2][4];
#pragma unroll
    for (int i = 0; i < 2; ++i)
#pragma unroll
        for (int j = 0; j < 4; ++j) acc[i][j] = (v8f){};

#pragma unroll 2
    for (int k = 0; k < DIN; k += 32) {
        __builtin_prefetch(arow0 + k + 64, 0, 3);
        v16bf a0 = load_frag(arow0, k, lh);
        v16bf a1 = load_frag(arow1, k, lh);
#pragma unroll
        for (int j = 0; j < 4; ++j) {
            v16bf bfr = load_frag(brow[j], k, lh);
            acc[0][j] = wmma_bf16(a0, bfr, acc[0][j]);
            acc[1][j] = wmma_bf16(a1, bfr, acc[1][j]);
        }
    }

    const int b      = m0 >> 11;          // /S
    const int s_base = m0 & (Sz - 1);
    const int h      = n0 >> 6;           // n0 is 64-aligned
    const size_t bh  = (size_t)(b * Hh + h);

    if (which == 2) {
#pragma unroll
        for (int i = 0; i < 2; ++i)
#pragma unroll
            for (int j = 0; j < 4; ++j) {
                int hd = 16 * j + row16;
                __bf16* col = Vtb + (bh * HD + hd) * Sz + s_base + 16 * i + (lh << 3);
#pragma unroll
                for (int r = 0; r < 8; ++r) col[r] = (__bf16)acc[i][j][r];
            }
    } else {
        __bf16* dst = (which == 0) ? Qb : Kb;
#pragma unroll
        for (int i = 0; i < 2; ++i)
#pragma unroll
            for (int r = 0; r < 8; ++r) {
                int srow = s_base + 16 * i + r + (lh << 3);
                __bf16* rowp = dst + (bh * Sz + srow) * HD + row16;
#pragma unroll
                for (int j = 0; j < 4; ++j) rowp[16 * j] = (__bf16)acc[i][j][r];
            }
    }
}

// ---------------------------------------------------------------------------
// Kernel 2: causal flash attention.  One wave owns a 16-query tile of one
// (b,h); streams 32-key blocks with online softmax.  K tiles are staged into
// LDS by the Tensor Data Mover (double-buffered, s_wait_tensorcnt), P is
// re-laid out C-frag -> A-frag through LDS (intra-wave, DS ops in-order).
// ---------------------------------------------------------------------------
__global__ __launch_bounds__(128) void flash_attn_kernel(
        const __bf16* __restrict__ Qb, const __bf16* __restrict__ Kb,
        const __bf16* __restrict__ Vtb, __bf16* __restrict__ ctxb) {
    __shared__ __align__(16) __bf16 Plds[4][16 * 32];
#if HAVE_TDM
    __shared__ __align__(16) __bf16 Klds[4][2][32 * HD];   // 32KB: double buffer
#endif

    const int wave  = threadIdx.x >> 5;
    const int lane  = threadIdx.x & 31;
    const int row16 = lane & 15;
    const int lh    = lane >> 4;

    const int qt = blockIdx.x * 4 + wave;   // query tile 0..127
    const int bh = blockIdx.y;              // 0..63  (b*H + h)
    const int q0 = qt * 16;

    const float scale = 0.125f;             // 1/sqrt(HD)

    // Preload Q fragments (HD=64 -> two K-steps of 32), reused for all blocks.
    const __bf16* Qrow = Qb + ((size_t)bh * Sz + q0 + row16) * HD;
    v16bf qf0 = load_frag(Qrow, 0, lh);
    v16bf qf1 = load_frag(Qrow, 32, lh);

    v8f acc[4] = {{}, {}, {}, {}};
    float mrow[8], lrow[8];
#pragma unroll
    for (int r = 0; r < 8; ++r) { mrow[r] = -3.0e38f; lrow[r] = 0.0f; }

    __bf16* P = Plds[wave];
    const int jend = q0 + 16;               // need keys < q0+16

#if HAVE_TDM
    const __bf16* Kbase = Kb + (size_t)bh * Sz * HD;
    // prime the pipeline: DMA first K tile into buffer 0
    tdm_load_k_tile((unsigned)(unsigned long long)(uintptr_t)(void*)&Klds[wave][0][0],
                    (unsigned long long)(uintptr_t)(const void*)Kbase);
    int buf = 0;
#endif

    for (int j0 = 0; j0 < jend; j0 += 32) {
#if HAVE_TDM
        __builtin_amdgcn_s_wait_tensorcnt(0);          // K tile for j0 landed
        if (j0 + 32 < jend)                            // overlap DMA(j0+32) w/ compute(j0)
            tdm_load_k_tile((unsigned)(unsigned long long)(uintptr_t)(void*)&Klds[wave][buf ^ 1][0],
                            (unsigned long long)(uintptr_t)(const void*)(Kbase + (size_t)(j0 + 32) * HD));
        const __bf16* Ksrc = &Klds[wave][buf][0];
        buf ^= 1;
#else
        const __bf16* Ksrc = Kb + ((size_t)bh * Sz + j0) * HD;
#endif
        const __bf16* Krow_lo = Ksrc + row16 * HD;
        const __bf16* Krow_hi = Krow_lo + 16 * HD;

        // ---- scores: S[16q x 32k] = Q (16x64) x K^T, two 16-key halves ----
        v8f slo = {}, shi = {};
        slo = wmma_bf16(qf0, load_frag(Krow_lo, 0, lh), slo);
        slo = wmma_bf16(qf1, load_frag(Krow_lo, 32, lh), slo);
        shi = wmma_bf16(qf0, load_frag(Krow_hi, 0, lh), shi);
        shi = wmma_bf16(qf1, load_frag(Krow_hi, 32, lh), shi);

        // ---- online softmax per row (rows live in half-wave groups) ----
#pragma unroll
        for (int r = 0; r < 8; ++r) {
            const int q   = q0 + r + (lh << 3);
            const int klo = j0 + row16;
            const int khi = klo + 16;
            float s0 = slo[r] * scale; if (klo > q) s0 = -3.0e38f;
            float s1 = shi[r] * scale; if (khi > q) s1 = -3.0e38f;

            float vmax = fmaxf(s0, s1);
#pragma unroll
            for (int off = 8; off >= 1; off >>= 1)
                vmax = fmaxf(vmax, __shfl_xor(vmax, off, 32));

            const float mnew  = fmaxf(mrow[r], vmax);
            const float alpha = __expf(mrow[r] - mnew);
            const float p0    = __expf(s0 - mnew);
            const float p1    = __expf(s1 - mnew);

            float rsum = p0 + p1;
#pragma unroll
            for (int off = 8; off >= 1; off >>= 1)
                rsum += __shfl_xor(rsum, off, 32);

            lrow[r] = lrow[r] * alpha + rsum;
            mrow[r] = mnew;
#pragma unroll
            for (int ss = 0; ss < 4; ++ss) acc[ss][r] *= alpha;

            const int prow = r + (lh << 3);
            P[prow * 32 + row16]      = (__bf16)p0;
            P[prow * 32 + row16 + 16] = (__bf16)p1;
        }

        // ---- acc += P (16x32) x V (32 x 64), four 16-col slices ----
        v16bf pf = load_frag(P + row16 * 32, 0, lh);   // DS in-order per wave
#pragma unroll
        for (int ss = 0; ss < 4; ++ss) {
            const __bf16* Vrow = Vtb + ((size_t)bh * HD + ss * 16 + row16) * Sz + j0;
            v16bf vf = load_frag(Vrow, 0, lh);
            acc[ss] = wmma_bf16(pf, vf, acc[ss]);
        }
    }

    // ---- normalize and store ctx (B,S,DOUT) bf16 ----
    const int b = bh >> 4;
    const int h = bh & (Hh - 1);
#pragma unroll
    for (int r = 0; r < 8; ++r) {
        const float inv = 1.0f / lrow[r];
        const int   q   = q0 + r + (lh << 3);
#pragma unroll
        for (int ss = 0; ss < 4; ++ss) {
            ctxb[((size_t)b * Sz + q) * DOUT + h * HD + ss * 16 + row16] =
                (__bf16)(acc[ss][r] * inv);
        }
    }
}

// ---------------------------------------------------------------------------
// Kernel 3: output projection  out = ctx @ Wo + bo  (f32 output), 32x64/wave.
// ---------------------------------------------------------------------------
__global__ __launch_bounds__(128) void out_proj_kernel(
        const __bf16* __restrict__ ctxb, const __bf16* __restrict__ Wot,
        const float* __restrict__ bo, float* __restrict__ out) {
    const int wave  = threadIdx.x >> 5;
    const int lane  = threadIdx.x & 31;
    const int row16 = lane & 15;
    const int lh    = lane >> 4;

    const int m0 = blockIdx.x * 32;
    const int n0 = blockIdx.y * 256 + wave * 64;

    const __bf16* arow0 = ctxb + (size_t)(m0 + row16) * DOUT;
    const __bf16* arow1 = arow0 + (size_t)16 * DOUT;
    const __bf16* brow[4];
#pragma unroll
    for (int j = 0; j < 4; ++j) brow[j] = Wot + (size_t)(n0 + 16 * j + row16) * DOUT;

    v8f acc[2][4];
#pragma unroll
    for (int i = 0; i < 2; ++i)
#pragma unroll
        for (int j = 0; j < 4; ++j) acc[i][j] = (v8f){};

#pragma unroll 2
    for (int k = 0; k < DOUT; k += 32) {
        __builtin_prefetch(arow0 + k + 64, 0, 3);
        v16bf a0 = load_frag(arow0, k, lh);
        v16bf a1 = load_frag(arow1, k, lh);
#pragma unroll
        for (int j = 0; j < 4; ++j) {
            v16bf bfr = load_frag(brow[j], k, lh);
            acc[0][j] = wmma_bf16(a0, bfr, acc[0][j]);
            acc[1][j] = wmma_bf16(a1, bfr, acc[1][j]);
        }
    }

#pragma unroll
    for (int i = 0; i < 2; ++i)
#pragma unroll
        for (int r = 0; r < 8; ++r) {
            int m = m0 + 16 * i + r + (lh << 3);
            float* rowp = out + (size_t)m * DOUT + n0 + row16;
#pragma unroll
            for (int j = 0; j < 4; ++j) rowp[16 * j] = acc[i][j][r] + bo[n0 + 16 * j + row16];
        }
}

// ---------------------------------------------------------------------------
// Launcher
// ---------------------------------------------------------------------------
extern "C" void kernel_launch(void* const* d_in, const int* in_sizes, int n_in,
                              void* d_out, int out_size, void* d_ws, size_t ws_size,
                              hipStream_t stream) {
    const float* x  = (const float*)d_in[0];
    const float* Wq = (const float*)d_in[1];
    const float* Wk = (const float*)d_in[2];
    const float* Wv = (const float*)d_in[3];
    const float* Wo = (const float*)d_in[4];
    const float* bo = (const float*)d_in[5];
    float* out = (float*)d_out;

    char* ws = (char*)d_ws;
    size_t off = 0;
    __bf16* xb  = (__bf16*)(ws + off); off += (size_t)MROWS * DIN * 2;        // 16 MB
    __bf16* Wqt = (__bf16*)(ws + off); off += (size_t)DIN * DOUT * 2;         // 2 MB
    __bf16* Wkt = (__bf16*)(ws + off); off += (size_t)DIN * DOUT * 2;
    __bf16* Wvt = (__bf16*)(ws + off); off += (size_t)DIN * DOUT * 2;
    __bf16* Wot = (__bf16*)(ws + off); off += (size_t)DOUT * DOUT * 2;
    __bf16* Qb  = (__bf16*)(ws + off); off += (size_t)MROWS * DOUT * 2;       // 16 MB
    __bf16* Kb  = (__bf16*)(ws + off); off += (size_t)MROWS * DOUT * 2;
    __bf16* Vtb = (__bf16*)(ws + off); off += (size_t)MROWS * DOUT * 2;
    __bf16* ctxb= (__bf16*)(ws + off); off += (size_t)MROWS * DOUT * 2;

    // 0) precision/layout conversion
    {
        int n = MROWS * DIN;
        cvt_x_kernel<<<(n + 255) / 256, 256, 0, stream>>>(x, xb, n);
        dim3 g((DIN * DOUT) / 256, 1, 4);
        cvt_wt_kernel<<<g, 256, 0, stream>>>(Wq, Wk, Wv, Wo, Wqt, Wkt, Wvt, Wot);
    }
    // 1) QKV projections (z = Q,K,V)
    {
        dim3 g(MROWS / 32, DOUT / 256, 3);
        proj_qkv_kernel<<<g, 128, 0, stream>>>(xb, Wqt, Wkt, Wvt, Qb, Kb, Vtb);
    }
    // 2) causal flash attention (TDM-staged K tiles)
    {
        dim3 g((Sz / 16) / 4, Bz * Hh, 1);
        flash_attn_kernel<<<g, 128, 0, stream>>>(Qb, Kb, Vtb, ctxb);
    }
    // 3) output projection + bias
    {
        dim3 g(MROWS / 32, DOUT / 256, 1);
        out_proj_kernel<<<g, 128, 0, stream>>>(ctxb, Wot, bo, out);
    }
}
